// ViT_17033840296597
// MI455X (gfx1250) — matrix operand order repaired and checked by reading, not compiled, not run on hardware
//
#include <hip/hip_runtime.h>
#include <hip/hip_bf16.h>
#include <math.h>

// ============================================================================
// LoFTR-style transformer (12 encoder layers, linear attention) for gfx1250.
// All dense math runs through v_wmma_f32_16x16x32_bf16 (f32 accumulate).
// Round 3: fused epilogues (elu+1, ReLU, bias, Z-scale), split-A GEMM to avoid
// concat materialization, fused LN+residual. Double-buffered LDS staging.
// ============================================================================

typedef __bf16 bf16_t;
typedef __attribute__((ext_vector_type(16))) __bf16 v16bf;
typedef __attribute__((ext_vector_type(8)))  __bf16 v8bf;
typedef __attribute__((ext_vector_type(8)))  float  v8f;

// ---------------------------------------------------------------------------
// WMMA LDS fragment loaders (CDNA5 16-bit A 16x32 / B 32x16 layouts, wave32)
// ---------------------------------------------------------------------------
__device__ __forceinline__ v16bf load_a_frag(const bf16_t* base, int row, int g, int ldk) {
  union { v16bf v; v8bf h[2]; } u;
  const bf16_t* p = base + row * ldk;
  u.h[0] = *(const v8bf*)(p + 16 * g);
  u.h[1] = *(const v8bf*)(p + 16 * g + 8);
  return u.v;
}
__device__ __forceinline__ v16bf load_b_frag(const bf16_t* base, int col, int g, int ldk) {
  union { v16bf v; v8bf h[2]; } u;
  const bf16_t* p = base + col * ldk + 16 * g;
  u.h[0] = *(const v8bf*)(p);
  u.h[1] = *(const v8bf*)(p + 8);
  return u.v;
}

// ---------------------------------------------------------------------------
// Generic GEMM: C[M,N] = [A | A2][M,K](f32) * B (bf16, pre-transposed [N][K])
//   A supplies k < Ksplit, A2 (optional) supplies k >= Ksplit (concat fusion).
// Tile 128x128x32, 256 threads = 8 waves, each wave 64x32 (4x2 WMMA accums).
// Double-buffered LDS, one barrier per K-step, global prefetch.
// Batched via blockIdx.z -> (nb, hb). Epilogue: optional zscale[row,h] mul,
// bias[n] add, act: 0=none 1=relu 2=elu+1.
// ---------------------------------------------------------------------------
#define GBM 128
#define GBN 128
#define GBK 32
#define GLDK 40   // 32 + 8 pad

__global__ void __launch_bounds__(256)
gemm_kernel(const float* __restrict__ A, long lda,
            const float* __restrict__ A2, long lda2, int Ksplit,
            const bf16_t* __restrict__ Bt, long ldb,
            float* __restrict__ C, long ldc,
            int M, int N, int K,
            long aBN, long aBH, long bBN, long bBH,
            long cBN, long cBH, int HB,
            const float* __restrict__ zscale, long zBN,
            const float* __restrict__ bias, int act)
{
  __shared__ bf16_t As[2 * GBM * GLDK];
  __shared__ bf16_t Bs[2 * GBN * GLDK];

  int z  = blockIdx.z;
  int nb = z / HB, hb = z - nb * HB;
  A  += (long)nb * aBN + (long)hb * aBH;
  Bt += (long)nb * bBN + (long)hb * bBH;
  C  += (long)nb * cBN + (long)hb * cBH;

  const int tid   = threadIdx.x;
  const int lane  = tid & 31;
  const int wave  = tid >> 5;
  const int waveM = wave >> 2;   // 0..1 -> 64-row half
  const int waveN = wave & 3;    // 0..3 -> 32-col quarter
  const int g     = lane >> 4;
  const int ln    = lane & 15;
  const int blockM = blockIdx.y * GBM;
  const int blockN = blockIdx.x * GBN;

  const int srow = tid >> 1;          // 0..127 (staging row)
  const int skc  = (tid & 1) * 16;    // 0 or 16

  const int ksp = A2 ? Ksplit : 0x7fffffff;
  const bool fast = (blockM + GBM <= M) && (blockN + GBN <= N) && ((K & (GBK - 1)) == 0);

  const float*  Ap  = A + (long)(blockM + srow) * lda;
  const float*  Ap2 = A2 ? (A2 + (long)(blockM + srow) * lda2) : nullptr;
  const bf16_t* Bp  = Bt + (long)(blockN + srow) * ldb + skc;
  bf16_t* asBase = As + srow * GLDK + skc;
  bf16_t* bsBase = Bs + srow * GLDK + skc;

  auto stage = [&](int buf, int k0) {
    bf16_t* as = asBase + buf * (GBM * GLDK);
    bf16_t* bs = bsBase + buf * (GBN * GLDK);
    if (fast) {
      int kk = k0 + skc;           // Ksplit is a multiple of 32 -> one source
      const float* asrc = (kk < ksp) ? (Ap + kk) : (Ap2 + (kk - Ksplit));
      const float4* ap = (const float4*)asrc;
      float4 f0 = ap[0], f1 = ap[1], f2 = ap[2], f3 = ap[3];
      const v8bf* bp = (const v8bf*)(Bp + k0);
      v8bf b0 = bp[0], b1 = bp[1];
      union { v8bf v; bf16_t e[8]; } u0, u1;
      u0.e[0] = (bf16_t)f0.x; u0.e[1] = (bf16_t)f0.y;
      u0.e[2] = (bf16_t)f0.z; u0.e[3] = (bf16_t)f0.w;
      u0.e[4] = (bf16_t)f1.x; u0.e[5] = (bf16_t)f1.y;
      u0.e[6] = (bf16_t)f1.z; u0.e[7] = (bf16_t)f1.w;
      u1.e[0] = (bf16_t)f2.x; u1.e[1] = (bf16_t)f2.y;
      u1.e[2] = (bf16_t)f2.z; u1.e[3] = (bf16_t)f2.w;
      u1.e[4] = (bf16_t)f3.x; u1.e[5] = (bf16_t)f3.y;
      u1.e[6] = (bf16_t)f3.z; u1.e[7] = (bf16_t)f3.w;
      *(v8bf*)(as)     = u0.v;
      *(v8bf*)(as + 8) = u1.v;
      *(v8bf*)(bs)     = b0;
      *(v8bf*)(bs + 8) = b1;
      __builtin_prefetch(asrc + 2 * GBK, 0, 0);
      __builtin_prefetch(Bp + k0 + 2 * GBK, 0, 0);
    } else {
      int gm = blockM + srow;
      int gn = blockN + srow;
      #pragma unroll 1
      for (int i = 0; i < 16; ++i) {
        int k = k0 + skc + i;
        float av = 0.0f;
        if (gm < M && k < K)
          av = (k < ksp) ? A[(long)gm * lda + k] : A2[(long)gm * lda2 + (k - Ksplit)];
        as[i] = (bf16_t)av;
        bf16_t bv = (gn < N && k < K) ? Bt[(long)gn * ldb + k] : (bf16_t)0.0f;
        bs[i] = bv;
      }
    }
  };

  v8f acc[4][2] = {};
  stage(0, 0);
  int buf = 0;
  for (int k0 = 0; k0 < K; k0 += GBK, buf ^= 1) {
    __syncthreads();                              // staging of `buf` complete
    if (k0 + GBK < K) stage(buf ^ 1, k0 + GBK);   // overlap next stage w/ math

    const bf16_t* as = As + buf * (GBM * GLDK);
    const bf16_t* bs = Bs + buf * (GBN * GLDK);
    v16bf afrag[4], bfrag[2];
    #pragma unroll
    for (int mt = 0; mt < 4; ++mt)
      afrag[mt] = load_a_frag(as, waveM * 64 + mt * 16 + ln, g, GLDK);
    #pragma unroll
    for (int nt = 0; nt < 2; ++nt)
      bfrag[nt] = load_b_frag(bs, waveN * 32 + nt * 16 + ln, g, GLDK);

    #pragma unroll
    for (int mt = 0; mt < 4; ++mt)
      #pragma unroll
      for (int nt = 0; nt < 2; ++nt)
        acc[mt][nt] = __builtin_amdgcn_wmma_f32_16x16x32_bf16(
            false, afrag[mt], false, bfrag[nt], (short)0, acc[mt][nt], false, false);
  }

  // Epilogue. C layout: VGPR i, lane group g -> m = i + 8g, n = lane&15.
  const float* zrow = zscale ? (zscale + (long)nb * zBN + hb) : nullptr;
  #pragma unroll
  for (int mt = 0; mt < 4; ++mt) {
    #pragma unroll
    for (int nt = 0; nt < 2; ++nt) {
      int gn = blockN + waveN * 32 + nt * 16 + ln;
      bool nok = fast || (gn < N);
      float badd = (bias && nok) ? bias[gn] : 0.0f;
      #pragma unroll
      for (int i = 0; i < 8; ++i) {
        int gm = blockM + waveM * 64 + mt * 16 + 8 * g + i;
        if (fast || (gm < M && nok)) {
          float v = acc[mt][nt][i];
          if (zrow) v *= zrow[(long)gm * 8];
          v += badd;
          if (act == 1) v = fmaxf(v, 0.0f);
          else if (act == 2) v = (v > 0.0f) ? (v + 1.0f) : __expf(v);
          C[(long)gm * ldc + gn] = v;
        }
      }
    }
  }
}

// ---------------------------------------------------------------------------
// KV = Kf^T * V  per (n,h): 64x64 output, reduction over S (split-K, atomics).
// grid = (numSlices, 64); block = 128 (4 waves, 2x2 of 32x32). Double-buffered.
// ---------------------------------------------------------------------------
#define KLDK 40

__global__ void __launch_bounds__(128)
kv_kernel(const float* __restrict__ Kf, const float* __restrict__ V,
          float* __restrict__ KV, int S, int sliceLen)
{
  __shared__ bf16_t Ks[2 * 64 * KLDK];   // [d][s]
  __shared__ bf16_t Vs[2 * 64 * KLDK];   // [v][s]

  int zb = blockIdx.y;
  int n = zb >> 3, h = zb & 7;
  const float* Kb = Kf + ((long)n * S) * 512 + h * 64;
  const float* Vb = V  + ((long)n * S) * 512 + h * 64;

  int s0   = blockIdx.x * sliceLen;
  int sEnd = s0 + sliceLen; if (sEnd > S) sEnd = S;

  const int tid  = threadIdx.x;
  const int lane = tid & 31, wave = tid >> 5;
  const int waveM = wave >> 1, waveN = wave & 1;
  const int g = lane >> 4, ln = lane & 15;

  auto stage = [&](int buf, int s) {
    bf16_t* ks = Ks + buf * (64 * KLDK);
    bf16_t* vs = Vs + buf * (64 * KLDK);
    #pragma unroll
    for (int it = 0; it < 4; ++it) {
      int idx = it * 128 + tid;     // 0..511
      int sl  = idx >> 4;           // 0..31
      int d4  = (idx & 15) * 4;     // 0..60
      int ss  = s + sl;
      float4 kf = {0.f, 0.f, 0.f, 0.f};
      float4 vf = {0.f, 0.f, 0.f, 0.f};
      if (ss < sEnd) {
        kf = *(const float4*)(Kb + (long)ss * 512 + d4);
        vf = *(const float4*)(Vb + (long)ss * 512 + d4);
      }
      ks[(d4 + 0) * KLDK + sl] = (bf16_t)kf.x;
      ks[(d4 + 1) * KLDK + sl] = (bf16_t)kf.y;
      ks[(d4 + 2) * KLDK + sl] = (bf16_t)kf.z;
      ks[(d4 + 3) * KLDK + sl] = (bf16_t)kf.w;
      vs[(d4 + 0) * KLDK + sl] = (bf16_t)vf.x;
      vs[(d4 + 1) * KLDK + sl] = (bf16_t)vf.y;
      vs[(d4 + 2) * KLDK + sl] = (bf16_t)vf.z;
      vs[(d4 + 3) * KLDK + sl] = (bf16_t)vf.w;
    }
  };

  v8f acc[2][2] = {};
  stage(0, s0);
  int buf = 0;
  for (int s = s0; s < sEnd; s += 32, buf ^= 1) {
    __syncthreads();
    if (s + 32 < sEnd) stage(buf ^ 1, s + 32);

    const bf16_t* ks = Ks + buf * (64 * KLDK);
    const bf16_t* vs = Vs + buf * (64 * KLDK);
    v16bf afrag[2], bfrag[2];
    #pragma unroll
    for (int mt = 0; mt < 2; ++mt)
      afrag[mt] = load_a_frag(ks, waveM * 32 + mt * 16 + ln, g, KLDK);
    #pragma unroll
    for (int nt = 0; nt < 2; ++nt)
      bfrag[nt] = load_b_frag(vs, waveN * 32 + nt * 16 + ln, g, KLDK);

    #pragma unroll
    for (int mt = 0; mt < 2; ++mt)
      #pragma unroll
      for (int nt = 0; nt < 2; ++nt)
        acc[mt][nt] = __builtin_amdgcn_wmma_f32_16x16x32_bf16(
            false, afrag[mt], false, bfrag[nt], (short)0, acc[mt][nt], false, false);
  }

  #pragma unroll
  for (int mt = 0; mt < 2; ++mt) {
    #pragma unroll
    for (int nt = 0; nt < 2; ++nt) {
      int vcol = waveN * 32 + nt * 16 + ln;
      #pragma unroll
      for (int i = 0; i < 8; ++i) {
        int d = waveM * 32 + mt * 16 + 8 * g + i;
        atomicAdd(&KV[(((long)zb * 64 + d) * 64) + vcol], acc[mt][nt][i]);
      }
    }
  }
}

// ---------------------------------------------------------------------------
// Glue kernels
// ---------------------------------------------------------------------------
__global__ void zero_kernel(float* p, long n) {
  for (long i = (long)blockIdx.x * 256 + threadIdx.x; i < n; i += (long)gridDim.x * 256)
    p[i] = 0.0f;
}
__global__ void tcvt_kernel(const float* __restrict__ W, bf16_t* __restrict__ Bt, int K, int N) {
  long total = (long)K * N;
  for (long i = (long)blockIdx.x * 256 + threadIdx.x; i < total; i += (long)gridDim.x * 256) {
    long k = i / N; int n = (int)(i - k * N);
    Bt[(long)n * K + k] = (bf16_t)W[i];
  }
}
__global__ void cvt_kernel(const float* __restrict__ W, bf16_t* __restrict__ B, long n) {
  for (long i = (long)blockIdx.x * 256 + threadIdx.x; i < n; i += (long)gridDim.x * 256)
    B[i] = (bf16_t)W[i];
}
__global__ void kvt_kernel(const float* __restrict__ KV, bf16_t* __restrict__ KVt) {
  long total = 64L * 64 * 64;
  for (long i = (long)blockIdx.x * 256 + threadIdx.x; i < total; i += (long)gridDim.x * 256) {
    long z = i >> 12; int r = (int)(i & 4095);
    int d = r >> 6, v = r & 63;
    KVt[(z * 64 + v) * 64 + d] = (bf16_t)KV[(z * 64 + d) * 64 + v];
  }
}
__global__ void ksum_kernel(const float* __restrict__ Kf, float* __restrict__ Ksum, int S) {
  int i = blockIdx.x * 256 + threadIdx.x;
  if (i >= 4096) return;
  int z = i >> 6, d = i & 63;
  int n = z >> 3, h = z & 7;
  const float* p = Kf + ((long)n * S) * 512 + h * 64 + d;
  float s = 0.0f;
  for (int ss = 0; ss < S; ++ss) s += p[(long)ss * 512];
  Ksum[i] = s;
}
__global__ void z_kernel(const float* __restrict__ Qf, const float* __restrict__ Ksum,
                         float* __restrict__ Z, int L) {
  long total = 8L * L * 8;
  for (long i = (long)blockIdx.x * 256 + threadIdx.x; i < total; i += (long)gridDim.x * 256) {
    int h = (int)(i & 7); long rl = i >> 3;          // rl = n*L + l
    int n = (int)(rl / L);
    const float* q  = Qf + rl * 512 + h * 64;
    const float* ks = Ksum + ((n << 3) + h) * 64;
    float s = 0.0f;
    #pragma unroll 8
    for (int d = 0; d < 64; ++d) s += q[d] * ks[d];
    Z[i] = 1.0f / (s + 1e-6f);
  }
}
// In-place LayerNorm over last dim (512), one block (256 thr) per row.
__global__ void __launch_bounds__(256)
ln512_kernel(float* __restrict__ x, const float* __restrict__ g,
             const float* __restrict__ b, long rows) {
  long row = blockIdx.x;
  if (row >= rows) return;
  float* p = x + row * 512;
  int t = threadIdx.x;
  float v0 = p[t], v1 = p[t + 256];
  __shared__ float red[256];
  red[t] = v0 + v1; __syncthreads();
  for (int s = 128; s > 0; s >>= 1) { if (t < s) red[t] += red[t + s]; __syncthreads(); }
  float mean = red[0] * (1.0f / 512.0f); __syncthreads();
  float d0 = v0 - mean, d1 = v1 - mean;
  red[t] = d0 * d0 + d1 * d1; __syncthreads();
  for (int s = 128; s > 0; s >>= 1) { if (t < s) red[t] += red[t + s]; __syncthreads(); }
  float rs = rsqrtf(red[0] * (1.0f / 512.0f) + 1e-5f);
  p[t]       = d0 * rs * g[t]       + b[t];
  p[t + 256] = d1 * rs * g[t + 256] + b[t + 256];
}
// x[row] += LN(src[row]) * g + b   (fused residual + LayerNorm)
__global__ void __launch_bounds__(256)
ln512_add_kernel(float* __restrict__ x, const float* __restrict__ src,
                 const float* __restrict__ g, const float* __restrict__ b, long rows) {
  long row = blockIdx.x;
  if (row >= rows) return;
  const float* p = src + row * 512;
  float* xp = x + row * 512;
  int t = threadIdx.x;
  float v0 = p[t], v1 = p[t + 256];
  __shared__ float red[256];
  red[t] = v0 + v1; __syncthreads();
  for (int s = 128; s > 0; s >>= 1) { if (t < s) red[t] += red[t + s]; __syncthreads(); }
  float mean = red[0] * (1.0f / 512.0f); __syncthreads();
  float d0 = v0 - mean, d1 = v1 - mean;
  red[t] = d0 * d0 + d1 * d1; __syncthreads();
  for (int s = 128; s > 0; s >>= 1) { if (t < s) red[t] += red[t + s]; __syncthreads(); }
  float rs = rsqrtf(red[0] * (1.0f / 512.0f) + 1e-5f);
  xp[t]       += d0 * rs * g[t]       + b[t];
  xp[t + 256] += d1 * rs * g[t + 256] + b[t + 256];
}
// conv_out + conv_b -> LN(pe) -> + pos_embed + vis_seg -> feat1 rows [0,3969)
__global__ void __launch_bounds__(256)
patch_finish_kernel(const float* __restrict__ conv, const float* __restrict__ conv_b,
                    const float* __restrict__ pe_g, const float* __restrict__ pe_b,
                    const float* __restrict__ pos, const float* __restrict__ vis,
                    float* __restrict__ feat1) {
  long r = blockIdx.x;                      // 0..31751
  int n = (int)(r / 3969), p = (int)(r - (long)n * 3969);
  const float* src = conv + r * 512;
  int t = threadIdx.x;
  float v0 = src[t] + conv_b[t], v1 = src[t + 256] + conv_b[t + 256];
  __shared__ float red[256];
  red[t] = v0 + v1; __syncthreads();
  for (int s = 128; s > 0; s >>= 1) { if (t < s) red[t] += red[t + s]; __syncthreads(); }
  float mean = red[0] * (1.0f / 512.0f); __syncthreads();
  float d0 = v0 - mean, d1 = v1 - mean;
  red[t] = d0 * d0 + d1 * d1; __syncthreads();
  for (int s = 128; s > 0; s >>= 1) { if (t < s) red[t] += red[t + s]; __syncthreads(); }
  float rs = rsqrtf(red[0] * (1.0f / 512.0f) + 1e-5f);
  float* dst = feat1 + ((long)n * 4096 + p) * 512;
  const float* pp = pos + (long)p * 512;
  dst[t]       = d0 * rs * pe_g[t]       + pe_b[t]       + pp[t]       + vis[t];
  dst[t + 256] = d1 * rs * pe_g[t + 256] + pe_b[t + 256] + pp[t + 256] + vis[t + 256];
}
// im2col for the image conv: A[8*3969, 1248]
__global__ void im2col_img_kernel(const float* __restrict__ img, float* __restrict__ out) {
  const long total = 31752L * 1248L;
  for (long i = (long)blockIdx.x * 256 + threadIdx.x; i < total; i += (long)gridDim.x * 256) {
    long r = i / 1248; int c = (int)(i - r * 1248);
    int n = (int)(r / 3969); int p = (int)(r - (long)n * 3969);
    int oh = p / 63, ow = p - oh * 63;
    int ci = c / 416; int rem = c - ci * 416;
    int kh = rem / 26, kw = rem - kh * 26;
    int ih = oh * 8 + kh, iw = ow * 13 + kw;
    out[i] = img[(((long)n * 3 + ci) * 512 + ih) * 832 + iw];
  }
}
// target -> [8, 6144] in (h, w, c) order
__global__ void build_a0_kernel(const float* __restrict__ target, float* __restrict__ out) {
  const long total = 8L * 6144;
  for (long i = (long)blockIdx.x * 256 + threadIdx.x; i < total; i += (long)gridDim.x * 256) {
    int n = (int)(i / 6144); int c6 = (int)(i - (long)n * 6144);
    int h = c6 / 192; int rem = c6 - h * 192;
    int w = rem / 3;  int ci = rem - w * 3;
    out[i] = target[(((long)n * 3 + ci) * 32 + h) * 64 + w];
  }
}
__global__ void tgt_finish_kernel(const float* __restrict__ tmp8, const float* __restrict__ bt,
                                  const float* __restrict__ vis, float* __restrict__ feat0) {
  int i = blockIdx.x * 256 + threadIdx.x;
  if (i >= 8 * 512) return;
  int n = i >> 9, c = i & 511;
  feat0[(long)n * 64 * 512 + c] = tmp8[i] + bt[c] + vis[c];
}
__global__ void feat0_text_kernel(const int* __restrict__ toks, const float* __restrict__ temb,
                                  const float* __restrict__ tseg, float* __restrict__ feat0) {
  const long total = 8L * 63 * 512;
  for (long i = (long)blockIdx.x * 256 + threadIdx.x; i < total; i += (long)gridDim.x * 256) {
    int c = (int)(i & 511); long r = i >> 9;
    int n = (int)(r / 63); int t = (int)(r - (long)n * 63);
    int tok = toks[n * 63 + t];
    feat0[((long)n * 64 + 1 + t) * 512 + c] = temb[(long)tok * 512 + c] + tseg[c];
  }
}
__global__ void feat1_text_kernel(const int* __restrict__ toks, const int* __restrict__ cxs,
                                  const int* __restrict__ cys, const float* __restrict__ temb,
                                  const float* __restrict__ cxe, const float* __restrict__ cye,
                                  const float* __restrict__ pid, const float* __restrict__ tseg,
                                  float* __restrict__ feat1) {
  const long total = 8L * 127 * 512;
  for (long i = (long)blockIdx.x * 256 + threadIdx.x; i < total; i += (long)gridDim.x * 256) {
    int c = (int)(i & 511); long r = i >> 9;
    int n = (int)(r / 127); int t = (int)(r - (long)n * 127);
    int tok = toks[n * 127 + t];
    int cx = cxs[n * 127 + t], cy = cys[n * 127 + t];
    feat1[((long)n * 4096 + 3969 + t) * 512 + c] =
        temb[(long)tok * 512 + c] + cxe[(long)cx * 512 + c] + cye[(long)cy * 512 + c] +
        pid[(long)t * 512 + c] + tseg[c];
  }
}
__global__ void meanpool_kernel(const float* __restrict__ feat1, float* __restrict__ pooled) {
  int i = blockIdx.x * 256 + threadIdx.x;
  if (i >= 8 * 512) return;
  int n = i >> 9, c = i & 511;
  const float* p = feat1 + ((long)n * 4096) * 512 + c;
  float s = 0.0f;
  for (int l = 0; l < 4096; ++l) s += p[(long)l * 512];
  pooled[i] = s * (1.0f / 4096.0f);
}
__global__ void head_final_kernel(const float* __restrict__ h2, const float* __restrict__ W,
                                  const float* __restrict__ bs, float* __restrict__ out) {
  int t = threadIdx.x;
  if (t >= 32) return;
  int n = t >> 2, j = t & 3;
  float s = bs[j];
  const float* x = h2 + n * 512;
  for (int d = 0; d < 512; ++d) s += x[d] * W[d * 4 + j];
  out[n * 4 + j] = 1.0f / (1.0f + __expf(-s));
}

// ---------------------------------------------------------------------------
// Host orchestration
// ---------------------------------------------------------------------------
static inline dim3 ewgrid(long n) {
  long b = (n + 255) / 256;
  if (b < 1) b = 1;
  if (b > 524288) b = 524288;
  return dim3((unsigned)b);
}

static void run_gemm(hipStream_t st, const float* A, long lda, const bf16_t* Bt, long ldb,
                     float* C, long ldc, long M, int N, int K,
                     const float* bias, int act,
                     const float* A2 = nullptr, long lda2 = 0, int Ksplit = 0,
                     int batches = 1, int HB = 1,
                     long aBN = 0, long aBH = 0, long bBN = 0, long bBH = 0,
                     long cBN = 0, long cBH = 0,
                     const float* zscale = nullptr, long zBN = 0) {
  dim3 grid((unsigned)((N + GBN - 1) / GBN), (unsigned)((M + GBM - 1) / GBM), (unsigned)batches);
  gemm_kernel<<<grid, 256, 0, st>>>(A, lda, A2, lda2, Ksplit, Bt, ldb, C, ldc, (int)M, N, K,
                                    aBN, aBH, bBN, bBH, cBN, cBH, HB, zscale, zBN, bias, act);
}

struct LayerW {
  const bf16_t *WqT, *WkT, *WvT, *WmT, *W1T, *W2T;
  const float *b1, *b2, *g1, *g2;
};
struct EncCtx {
  hipStream_t st;
  float *bufQ, *bufK, *bufV, *bufO, *bufM, *bufH, *KV, *Ksum, *Zbuf;
  bf16_t* KVt;
};

static void encoder(const EncCtx& c, float* x, int L, const float* src, int S, const LayerW& w) {
  hipStream_t st = c.st;
  long ML = 8L * L, MS = 8L * S;

  // Qf = elu(x@Wq)+1, Kf = elu(src@Wk)+1 (fused epilogue), V = src@Wv
  run_gemm(st, x,   512, w.WqT, 512, c.bufQ, 512, ML, 512, 512, nullptr, 2);
  run_gemm(st, src, 512, w.WkT, 512, c.bufK, 512, MS, 512, 512, nullptr, 2);
  run_gemm(st, src, 512, w.WvT, 512, c.bufV, 512, MS, 512, 512, nullptr, 0);

  zero_kernel<<<ewgrid(64L * 64 * 64), 256, 0, st>>>(c.KV, 64L * 64 * 64);
  {
    int slices = (S + 511) / 512;
    dim3 grid((unsigned)slices, 64);
    kv_kernel<<<grid, 128, 0, st>>>(c.bufK, c.bufV, c.KV, S, 512);
  }
  ksum_kernel<<<16, 256, 0, st>>>(c.bufK, c.Ksum, S);
  kvt_kernel<<<ewgrid(64L * 64 * 64), 256, 0, st>>>(c.KV, c.KVt);
  z_kernel<<<ewgrid(8L * L * 8), 256, 0, st>>>(c.bufQ, c.Ksum, c.Zbuf, L);

  // out[l, h*64+v] = (Qf_head . KV_head) * Z[l,h]  (Z fused into epilogue)
  run_gemm(st, c.bufQ, 512, c.KVt, 64, c.bufO, 512,
           L, 64, 64, nullptr, 0,
           nullptr, 0, 0,
           64, 8,
           (long)L * 512, 64,           // A: per-n, per-h strides
           8L * 64 * 64, 64L * 64,      // B (KVt): per-n, per-h
           (long)L * 512, 64,           // C: per-n, per-h
           c.Zbuf, (long)L * 8);        // zscale strides
  run_gemm(st, c.bufO, 512, w.WmT, 512, c.bufM, 512, ML, 512, 512, nullptr, 0);
  ln512_kernel<<<(unsigned)ML, 256, 0, st>>>(c.bufM, w.g1, w.b1, ML);

  // FFN with fused concat: [x | msg] @ W1 -> relu -> @ W2 -> LN -> += x
  run_gemm(st, x, 512, w.W1T, 1024, c.bufH, 1024, ML, 1024, 1024, nullptr, 1,
           c.bufM, 512, 512);
  run_gemm(st, c.bufH, 1024, w.W2T, 1024, c.bufQ, 512, ML, 512, 1024, nullptr, 0);
  ln512_add_kernel<<<(unsigned)ML, 256, 0, st>>>(x, c.bufQ, w.g2, w.b2, ML);
}

extern "C" void kernel_launch(void* const* d_in, const int* in_sizes, int n_in,
                              void* d_out, int out_size, void* d_ws, size_t ws_size,
                              hipStream_t stream) {
  (void)in_sizes; (void)n_in; (void)out_size; (void)ws_size;
  hipStream_t st = stream;

  // ---- inputs (jax pytree / sorted-dict-key order) ----
  const float* img          = (const float*)d_in[0];
  const int*   img_texts    = (const int*)d_in[1];
  const int*   c_xs         = (const int*)d_in[2];
  const int*   c_ys         = (const int*)d_in[3];
  const float* target       = (const float*)d_in[4];
  const int*   target_texts = (const int*)d_in[5];
  const float* Wt        = (const float*)d_in[6];
  const float* bt        = (const float*)d_in[7];
  const float* conv_b    = (const float*)d_in[8];
  const float* conv_w    = (const float*)d_in[9];
  const float* cx_emb    = (const float*)d_in[10];
  const float* cy_emb    = (const float*)d_in[11];
  const float* head_W0   = (const float*)d_in[12];
  const float* head_W1   = (const float*)d_in[13];
  const float* head_W2   = (const float*)d_in[14];
  const float* head_b    = (const float*)d_in[15];
  const float* head_bs0  = (const float*)d_in[16];
  const float* head_bs1  = (const float*)d_in[17];
  const float* head_bs2  = (const float*)d_in[18];
  const float* head_g    = (const float*)d_in[19];
  // layers at 20 + 10*i: W1,W2,Wk,Wm,Wq,Wv,b1,b2,g1,g2 (sorted keys)
  const float* pe_b      = (const float*)d_in[140];
  const float* pe_g      = (const float*)d_in[141];
  const float* pos_embed = (const float*)d_in[142];
  const float* posid_emb = (const float*)d_in[143];
  const float* text_emb  = (const float*)d_in[144];
  const float* txt_seg   = (const float*)d_in[145];
  const float* vis_seg   = (const float*)d_in[146];
  float* out = (float*)d_out;

  // ---- workspace layout ----
  char* cur = (char*)d_ws;
  auto take = [&](size_t bytes) -> char* {
    char* p = cur;
    cur += (bytes + 255) & ~(size_t)255;
    return p;
  };
  float* feat0  = (float*)take(8L * 64 * 512 * 4);
  float* feat1  = (float*)take(8L * 4096 * 512 * 4);
  float* bufQ   = (float*)take(8L * 4096 * 512 * 4);
  float* bufK   = (float*)take(8L * 4096 * 512 * 4);
  float* bufV   = (float*)take(8L * 4096 * 512 * 4);
  float* bufO   = (float*)take(8L * 4096 * 512 * 4);
  float* bufM   = (float*)take(8L * 4096 * 512 * 4);
  float* bufH   = (float*)take(8L * 4096 * 1024 * 4);
  float* imcol  = (float*)take(31752L * 1248 * 4);
  float* KV     = (float*)take(64L * 64 * 64 * 4);
  bf16_t* KVt   = (bf16_t*)take(64L * 64 * 64 * 2);
  float* Ksum   = (float*)take(4096 * 4);
  float* Zbuf   = (float*)take(8L * 4096 * 8 * 4);
  float* A0     = (float*)take(8L * 6144 * 4);
  float* tmp8   = (float*)take(8L * 512 * 4);
  float* pooled = (float*)take(8L * 512 * 4);
  float* h1     = (float*)take(8L * 512 * 4);
  float* h2     = (float*)take(8L * 512 * 4);
  bf16_t* WtT    = (bf16_t*)take(512L * 6144 * 2);
  bf16_t* convWb = (bf16_t*)take(512L * 1248 * 2);
  bf16_t* hw0t   = (bf16_t*)take(512L * 512 * 2);
  bf16_t* hw1t   = (bf16_t*)take(512L * 512 * 2);
  LayerW lw[12];
  for (int i = 0; i < 12; ++i) {
    lw[i].WqT = (bf16_t*)take(512L * 512 * 2);
    lw[i].WkT = (bf16_t*)take(512L * 512 * 2);
    lw[i].WvT = (bf16_t*)take(512L * 512 * 2);
    lw[i].WmT = (bf16_t*)take(512L * 512 * 2);
    lw[i].W1T = (bf16_t*)take(1024L * 1024 * 2);
    lw[i].W2T = (bf16_t*)take(512L * 1024 * 2);
    int base = 20 + 10 * i;
    lw[i].b1 = (const float*)d_in[base + 6];
    lw[i].b2 = (const float*)d_in[base + 7];
    lw[i].g1 = (const float*)d_in[base + 8];
    lw[i].g2 = (const float*)d_in[base + 9];
  }

  // ---- weight prep: convert / transpose to bf16 [N][K] ----
  tcvt_kernel<<<ewgrid(6144L * 512), 256, 0, st>>>(Wt, WtT, 6144, 512);
  cvt_kernel<<<ewgrid(512L * 1248), 256, 0, st>>>(conv_w, convWb, 512L * 1248);
  tcvt_kernel<<<ewgrid(512L * 512), 256, 0, st>>>(head_W0, hw0t, 512, 512);
  tcvt_kernel<<<ewgrid(512L * 512), 256, 0, st>>>(head_W1, hw1t, 512, 512);
  for (int i = 0; i < 12; ++i) {
    int base = 20 + 10 * i;
    tcvt_kernel<<<ewgrid(512L * 512), 256, 0, st>>>((const float*)d_in[base + 4], (bf16_t*)lw[i].WqT, 512, 512);
    tcvt_kernel<<<ewgrid(512L * 512), 256, 0, st>>>((const float*)d_in[base + 2], (bf16_t*)lw[i].WkT, 512, 512);
    tcvt_kernel<<<ewgrid(512L * 512), 256, 0, st>>>((const float*)d_in[base + 5], (bf16_t*)lw[i].WvT, 512, 512);
    tcvt_kernel<<<ewgrid(512L * 512), 256, 0, st>>>((const float*)d_in[base + 3], (bf16_t*)lw[i].WmT, 512, 512);
    tcvt_kernel<<<ewgrid(1024L * 1024), 256, 0, st>>>((const float*)d_in[base + 0], (bf16_t*)lw[i].W1T, 1024, 1024);
    tcvt_kernel<<<ewgrid(1024L * 512), 256, 0, st>>>((const float*)d_in[base + 1], (bf16_t*)lw[i].W2T, 1024, 512);
  }

  // ---- feat0: target linear + text tokens ----
  build_a0_kernel<<<ewgrid(8L * 6144), 256, 0, st>>>(target, A0);
  run_gemm(st, A0, 6144, WtT, 6144, tmp8, 512, 8, 512, 6144, nullptr, 0);
  tgt_finish_kernel<<<16, 256, 0, st>>>(tmp8, bt, vis_seg, feat0);
  feat0_text_kernel<<<ewgrid(8L * 63 * 512), 256, 0, st>>>(target_texts, text_emb, txt_seg, feat0);

  // ---- feat1: conv patch embed (im2col + WMMA GEMM) + text tokens ----
  im2col_img_kernel<<<ewgrid(31752L * 1248), 256, 0, st>>>(img, imcol);
  run_gemm(st, imcol, 1248, convWb, 1248, bufO, 512, 31752, 512, 1248, nullptr, 0);
  patch_finish_kernel<<<31752, 256, 0, st>>>(bufO, conv_b, pe_g, pe_b, pos_embed, vis_seg, feat1);
  feat1_text_kernel<<<ewgrid(8L * 127 * 512), 256, 0, st>>>(
      img_texts, c_xs, c_ys, text_emb, cx_emb, cy_emb, posid_emb, txt_seg, feat1);

  // ---- transformer: alternating self / cross ----
  EncCtx c = {st, bufQ, bufK, bufV, bufO, bufM, bufH, KV, Ksum, Zbuf, KVt};
  for (int i = 0; i < 12; ++i) {
    if ((i & 1) == 0) {
      encoder(c, feat0, 64, feat0, 64, lw[i]);
      encoder(c, feat1, 4096, feat1, 4096, lw[i]);
    } else {
      encoder(c, feat0, 64, feat1, 4096, lw[i]);
      encoder(c, feat1, 4096, feat0, 64, lw[i]);   // uses updated feat0
    }
  }

  // ---- head ----
  meanpool_kernel<<<16, 256, 0, st>>>(feat1, pooled);
  ln512_kernel<<<8, 256, 0, st>>>(pooled, head_g, head_b, 8);
  run_gemm(st, pooled, 512, hw0t, 512, h1, 512, 8, 512, 512, head_bs0, 1);
  run_gemm(st, h1,     512, hw1t, 512, h2, 512, 8, 512, 512, head_bs1, 1);
  head_final_kernel<<<1, 32, 0, st>>>(h2, head_W2, head_bs2, out);
}
